// Our_8237747274084
// MI455X (gfx1250) — matrix operand order repaired and checked by reading, not compiled, not run on hardware
//
#include <hip/hip_runtime.h>
#include <hip/hip_bf16.h>

typedef __attribute__((ext_vector_type(16))) _Float16 v16h;
typedef __attribute__((ext_vector_type(8)))  float    v8f;

#define N_NODES 100000
#define NFEAT   128
#define NHID    128
#define NCLASS  8
#define BN_EPS  1e-5f

__device__ __forceinline__ void atomAddF(float* p, float v) {
    __hip_atomic_fetch_add(p, v, __ATOMIC_RELAXED, __HIP_MEMORY_SCOPE_AGENT);
}

// ---------------------------------------------------------------------------
// 1) deg[n] = 1 (self loop); zero BN stats scratch
__global__ void k_init(float* deg, float* stats /*512 floats*/) {
    int i = blockIdx.x * blockDim.x + threadIdx.x;
    if (i < N_NODES) deg[i] = 1.0f;
    if (i < 512) stats[i] = 0.0f;
}

// 2) deg[dst] += 1 over edges
__global__ void k_deg(const int* __restrict__ ei, float* deg, int E) {
    int i = blockIdx.x * blockDim.x + threadIdx.x;
    int stride = gridDim.x * blockDim.x;
    for (int e = i; e < E; e += stride) {
        int d = ei[E + e];
        atomAddF(&deg[d], 1.0f);
    }
}

// 3) dinv[n] = rsqrt(deg[n])   (deg >= 1 always, self-loop)
__global__ void k_dinv(float* dinv) {
    int i = blockIdx.x * blockDim.x + threadIdx.x;
    if (i < N_NODES) dinv[i] = __frsqrt_rn(dinv[i]);
}

// ---------------------------------------------------------------------------
// 4) h = f16(x) @ f16(W) + b_gc  via v_wmma_f32_16x16x32_f16.
//    Block = 256 threads = 8 wave32; block computes a 16x128 strip of h.
//    Wave w computes the 16x16 tile at columns [16w, 16w+16).
//    W is staged in LDS PRE-PACKED in the B-fragment register layout, so each
//    lane's fragment is 16 contiguous halves -> two ds_load_b128 per fragment.
__global__ void k_gemm(const float* __restrict__ x,
                       const float* __restrict__ W,
                       const float* __restrict__ b_gc,
                       float* __restrict__ h) {
    // layout: [tile(8)][kc(4)][lane(32)][slot(16)] halves = 32 KB
    __shared__ __align__(32) _Float16 Wlds[NFEAT * NHID];
    const int tid  = threadIdx.x;
    const int lane = tid & 31;
    const int wave = tid >> 5;           // 0..7 -> column tile
    const int half = lane >> 4;          // 0 or 1
    const int l16  = lane & 15;

    // cooperative W -> LDS: coalesced global read (consecutive tid = consecutive n),
    // scatter on the LDS-write side into fragment order.
    for (int i = tid; i < NFEAT * NHID; i += 256) {
        int k = i >> 7, n = i & 127;
        int tile = n >> 4;
        int kc   = k >> 5;
        int flane = ((k >> 4) & 1) * 16 + (n & 15);
        int slot  = k & 15;
        Wlds[(((tile * 4 + kc) * 32 + flane) << 4) + slot] = (_Float16)W[i];
    }
    __syncthreads();

    const int m0 = blockIdx.x * 16;          // 100000 % 16 == 0
    const int m  = m0 + l16;                 // A-row this lane supplies
    const int nc = wave * 16 + l16;          // D column this lane owns
    const float* xrow = x + (size_t)m * NFEAT;
    const _Float16* bfrag_base = Wlds + (((wave * 4) * 32 + lane) << 4);

    v8f acc = {};
    #pragma unroll
    for (int kc = 0; kc < 4; ++kc) {
        const int k0 = kc * 32;
        // A fragment: VGPR v holds K = kbase, kbase+1 (contiguous k = 16*half + s, same order as B);
        // one contiguous 16-float run per lane -> global_load_b128 + v_cvt_pk.
        v16h a;
        #pragma unroll
        for (int v = 0; v < 8; ++v) {
            int kbase = 2 * v + half * 16;
            float2 p = *(const float2*)(xrow + k0 + kbase);
            a[2 * v]     = (_Float16)p.x;
            a[2 * v + 1] = (_Float16)p.y;
        }
        // B fragment: one aligned 32-byte LDS read (2x ds_load_b128)
        v16h b = *(const v16h*)(bfrag_base + (kc << 9));
        acc = __builtin_amdgcn_wmma_f32_16x16x32_f16(
            false, a, false, b, (short)0, acc, false, false);
    }

    const float bias = b_gc[nc];
    #pragma unroll
    for (int r = 0; r < 8; ++r) {
        int mrow = m0 + r + half * 8;        // D layout: VGPR r -> row r (+8 for hi lanes)
        h[(size_t)mrow * NHID + nc] = acc[r] + bias;
    }
}

// ---------------------------------------------------------------------------
// 5) self-loop contribution: agg[n] = h[n] * dinv[n]^2   (norm = dinv*dinv)
__global__ void k_selfloop(const float* __restrict__ h,
                           const float* __restrict__ dinv,
                           float* __restrict__ agg) {
    size_t i = (size_t)blockIdx.x * blockDim.x + threadIdx.x;
    if (i < (size_t)N_NODES * NHID) {
        int n = (int)(i >> 7);
        float di = dinv[n];
        agg[i] = h[i] * di * di;
    }
}

// 6) edge scatter: one wave32 per edge; 32 lanes cover the 128-float row
//    as float4, gathered coalesced, scattered with f32 global atomics.
__global__ void k_scatter(const int* __restrict__ ei,
                          const float* __restrict__ h,
                          const float* __restrict__ dinv,
                          float* __restrict__ agg, int E) {
    const int lane = threadIdx.x & 31;
    const int wave = threadIdx.x >> 5;
    int e = blockIdx.x * 8 + wave;
    const int stride = gridDim.x * 8;
    for (; e < E; e += stride) {
        int s = ei[e];
        int d = ei[E + e];
        float w = dinv[s] * dinv[d];
        float4 v = ((const float4*)(h + (size_t)s * NHID))[lane];
        float* o = agg + (size_t)d * NHID + lane * 4;
        atomAddF(o + 0, v.x * w);
        atomAddF(o + 1, v.y * w);
        atomAddF(o + 2, v.z * w);
        atomAddF(o + 3, v.w * w);
    }
}

// ---------------------------------------------------------------------------
// 7) BN partial sums: each thread owns feature f = tid&127, strides rows,
//    one atomic pair per thread, two per (block, feature).
__global__ void k_bnsums(const float* __restrict__ agg,
                         float* __restrict__ sums,
                         float* __restrict__ sumsq) {
    const int f   = threadIdx.x & 127;
    const int row = threadIdx.x >> 7;     // 0 or 1
    float s = 0.f, ss = 0.f;
    for (int r = blockIdx.x * 2 + row; r < N_NODES; r += gridDim.x * 2) {
        float v = agg[(size_t)r * NHID + f];
        s += v; ss += v * v;
    }
    atomAddF(&sums[f], s);
    atomAddF(&sumsq[f], ss);
}

// 8) mean / rstd
__global__ void k_bnfin(const float* __restrict__ sums,
                        const float* __restrict__ sumsq,
                        float* __restrict__ mean,
                        float* __restrict__ rstd) {
    int f = threadIdx.x;
    if (f < NHID) {
        float m = sums[f] * (1.0f / N_NODES);
        float v = sumsq[f] * (1.0f / N_NODES) - m * m;
        mean[f] = m;
        rstd[f] = __frsqrt_rn(v + BN_EPS);
    }
}

// 9) in-place normalize: embed = gamma*(agg-mean)*rstd + beta
__global__ void k_bnapply(float* __restrict__ embed,
                          const float* __restrict__ mean,
                          const float* __restrict__ rstd,
                          const float* __restrict__ gamma,
                          const float* __restrict__ beta) {
    size_t i = (size_t)blockIdx.x * blockDim.x + threadIdx.x;
    if (i < (size_t)N_NODES * NHID) {
        int f = (int)(i & 127);
        embed[i] = gamma[f] * (embed[i] - mean[f]) * rstd[f] + beta[f];
    }
}

// 10) FC head: out[n][c] = sum_k<64 embed[n][k] * fc_w[c][k] + fc_b[c]
__global__ void k_fc(const float* __restrict__ embed,
                     const float* __restrict__ fc_w,
                     const float* __restrict__ fc_b,
                     float* __restrict__ out) {
    __shared__ float wfc[NCLASS * 64];
    __shared__ float bfc[NCLASS];
    const int tid = threadIdx.x;
    for (int i = tid; i < NCLASS * 64; i += 256) wfc[i] = fc_w[i];
    if (tid < NCLASS) bfc[tid] = fc_b[tid];
    __syncthreads();
    int idx = blockIdx.x * 256 + tid;
    if (idx < N_NODES * NCLASS) {
        int n = idx >> 3, c = idx & 7;
        const float* er = embed + (size_t)n * NHID;
        const float* wr = wfc + c * 64;
        float s = bfc[c];
        #pragma unroll 8
        for (int k = 0; k < 64; ++k) s += er[k] * wr[k];
        out[idx] = s;
    }
}

// ---------------------------------------------------------------------------
extern "C" void kernel_launch(void* const* d_in, const int* in_sizes, int n_in,
                              void* d_out, int out_size, void* d_ws, size_t ws_size,
                              hipStream_t stream) {
    const float* x     = (const float*)d_in[0];
    const int*   ei    = (const int*)  d_in[1];
    const float* W     = (const float*)d_in[2];
    const float* b_gc  = (const float*)d_in[3];
    const float* gamma = (const float*)d_in[4];
    const float* beta  = (const float*)d_in[5];
    const float* fc_w  = (const float*)d_in[6];
    const float* fc_b  = (const float*)d_in[7];
    const int E = in_sizes[1] / 2;

    // d_out layout: out [100000*8] then embed [100000*128]
    float* out   = (float*)d_out;
    float* embed = out + (size_t)N_NODES * NCLASS;  // also the aggregation buffer

    // workspace carve (floats): dinv | h | sums | sumsq | mean | rstd
    float* dinv  = (float*)d_ws;
    float* h     = dinv + N_NODES;                    // N_NODES % 16 == 0, aligned
    float* stats = h + (size_t)N_NODES * NHID;        // 512 floats
    float* sums  = stats;
    float* sumsq = stats + 128;
    float* mean  = stats + 256;
    float* rstd  = stats + 384;

    const int T = 256;
    k_init    <<<(N_NODES + T - 1) / T, T, 0, stream>>>(dinv, stats);
    k_deg     <<<2048, T, 0, stream>>>(ei, dinv, E);
    k_dinv    <<<(N_NODES + T - 1) / T, T, 0, stream>>>(dinv);
    k_gemm    <<<N_NODES / 16, T, 0, stream>>>(x, W, b_gc, h);
    k_selfloop<<<(N_NODES * NHID + T - 1) / T, T, 0, stream>>>(h, dinv, embed);
    k_scatter <<<4096, T, 0, stream>>>(ei, h, dinv, embed, E);
    k_bnsums  <<<1024, T, 0, stream>>>(embed, sums, sumsq);
    k_bnfin   <<<1, 128, 0, stream>>>(sums, sumsq, mean, rstd);
    k_bnapply <<<(N_NODES * NHID + T - 1) / T, T, 0, stream>>>(embed, mean, rstd, gamma, beta);
    k_fc      <<<(N_NODES * NCLASS + T - 1) / T, T, 0, stream>>>(embed, fc_w, fc_b, out);
}
